// VertexConv_22204980920649
// MI455X (gfx1250) — compile-verified, matrix operand order checked by reading, not run on hardware
//
#include <hip/hip_runtime.h>
#include <hip/hip_bf16.h>
#include <stdint.h>

// ---------------- problem constants ----------------
#define B_   4
#define M_   2048
#define E_   1024
#define D_   256
#define H_   8
#define DH_  32
#define TD_  768     // 3*D
#define OUT_ 256

typedef __attribute__((ext_vector_type(16))) _Float16 v16h;
typedef __attribute__((ext_vector_type(8)))  float    v8f;
typedef int v4i __attribute__((vector_size(16)));

union HV { v16h h; unsigned u[8]; };

// D = A(16x32 f16) * B(32x16 f16) + C(16x16 f32)
__device__ __forceinline__ v8f wmma16(const HV& a, const HV& b, v8f c) {
  return __builtin_amdgcn_wmma_f32_16x16x32_f16(
      /*neg_a=*/false, a.h, /*neg_b=*/false, b.h,
      /*c_mod=*/(short)0, c, /*reuse_a=*/false, /*reuse_b=*/false);
}

// Load one A/B fragment from global. `p` points at this lane's row base
// (row*ld + (lane>=16 ? 8 : 0)). K pairs per dword follow the gfx1250
// 16-bit layout: vgpr v covers k = (v>>2)*16 + (v&3)*2 (+8 for hi half).
__device__ __forceinline__ void load_frag(HV& f, const _Float16* p) {
#pragma unroll
  for (int v = 0; v < 8; ++v) {
    const int koff = ((v >> 2) << 4) + ((v & 3) << 1);
    f.u[v] = *(const unsigned*)(p + koff);
  }
}

#define HAS_ASYNC_LDS __has_builtin(__builtin_amdgcn_global_load_async_to_lds_b128)

#if HAS_ASYNC_LDS
typedef __attribute__((address_space(3))) const _Float16 lds_ch;

// one lane moves 16 bytes global -> LDS, no destination VGPRs (ASYNCcnt)
#define ASYNC_CP16(gsrc, ldst)                                        \
  __builtin_amdgcn_global_load_async_to_lds_b128(                     \
      (v4i*)(uintptr_t)(gsrc), (v4i*)(ldst), 0, 0)

template <int N>
__device__ __forceinline__ void wait_async() {
#if __has_builtin(__builtin_amdgcn_s_wait_asynccnt)
  __builtin_amdgcn_s_wait_asynccnt((unsigned short)N);
#else
  asm volatile("s_wait_asynccnt %0" ::"i"(N) : "memory");
#endif
  asm volatile("" ::: "memory");
}

// fragment read out of LDS (2x ds_load_b128 per fragment)
__device__ __forceinline__ void load_frag_lds(HV& f, lds_ch* p) {
#pragma unroll
  for (int v = 0; v < 8; ++v) {
    const int koff = ((v >> 2) << 4) + ((v & 3) << 1);
    f.u[v] = *(__attribute__((address_space(3))) const unsigned*)(p + koff);
  }
}
#endif

// C[16 x NT*16] = A[16xK] * (Bt tiles)^T.
// Async-LDS pipelined: stage k-step tiles into a per-wave double buffer with
// GLOBAL_LOAD_ASYNC_TO_LDS_B128, overlap with WMMA on the previous buffer.
// Block size for callers of this path must be 128 (4 waves). K % 32 == 0.
template <int NT>
__device__ __forceinline__ void gemm_tileN(const _Float16* A, int lda,
                                           const _Float16* Bt, int ldb,
                                           int K, v8f* acc) {
  const int lane = threadIdx.x & 31;
  const int half = lane >> 4, idx = lane & 15;
#if HAS_ASYNC_LDS
  constexpr int TILE = 16 * 32;                 // halves per 16x32 tile
  constexpr int BUF  = (1 + NT) * TILE;         // halves per k-step buffer
  __shared__ _Float16 smem[4][2 * BUF];         // 4 waves per block
  _Float16* my = smem[threadIdx.x >> 5];

  const int srow = lane >> 2;                   // 8 rows per async op
  const int scol = (lane & 3) * 8;              // halves (16B chunk) in a row

  auto stage = [&](int k0, int buf) {
    _Float16* dst = my + buf * BUF;
#pragma unroll
    for (int rr = 0; rr < 2; ++rr) {            // A tile: rows 0..7, 8..15
      const _Float16* g = A + (size_t)(srow + rr * 8) * lda + k0 + scol;
      ASYNC_CP16(g, dst + (srow + rr * 8) * 32 + scol);
    }
#pragma unroll
    for (int t = 0; t < NT; ++t)
#pragma unroll
      for (int rr = 0; rr < 2; ++rr) {
        const _Float16* g = Bt + (size_t)(t * 16 + srow + rr * 8) * ldb + k0 + scol;
        ASYNC_CP16(g, dst + (1 + t) * TILE + (srow + rr * 8) * 32 + scol);
      }
  };

  // all fragments of a buffer preloaded (one ds clause), then the WMMA chain
  auto compute = [&](int buf) {
    lds_ch* src = (lds_ch*)(my + buf * BUF);
    HV a;
    load_frag_lds(a, src + idx * 32 + half * 8);
    HV b[NT];
#pragma unroll
    for (int t = 0; t < NT; ++t)
      load_frag_lds(b[t], src + (1 + t) * TILE + idx * 32 + half * 8);
#pragma unroll
    for (int t = 0; t < NT; ++t) acc[t] = wmma16(a, b[t], acc[t]);
  };

  stage(0, 0);
  int buf = 0;
  int k0 = 0;
  for (; k0 < K - 32; k0 += 32) {               // steady state: no branches
    stage(k0 + 32, buf ^ 1);                    // next stage in flight...
    wait_async<2 * (1 + NT)>();                 // ...current stage complete
    compute(buf);
    buf ^= 1;
  }
  wait_async<0>();                              // peeled last step
  compute(buf);
#else
  // fallback: direct global fragment loads
  const _Float16* ar = A  + (size_t)idx * lda + half * 8;
  const _Float16* br = Bt + (size_t)idx * ldb + half * 8;
  for (int k0 = 0; k0 < K; k0 += 32) {
    HV a;
    load_frag(a, ar + k0);
#pragma unroll
    for (int t = 0; t < NT; ++t) {
      HV b;
      load_frag(b, br + (size_t)t * 16 * ldb + k0);
      acc[t] = wmma16(a, b, acc[t]);
    }
  }
#endif
}

// ---------------- conversion / repack kernels ----------------
__global__ void k_f32_to_f16(const float* __restrict__ in,
                             _Float16* __restrict__ out, int n) {
  int i = blockIdx.x * blockDim.x + threadIdx.x;
  if (i < n) out[i] = (_Float16)in[i];
}

// inc [b,m,e] f32 -> incT [b,e,m] f16 via 32x32 LDS tiles (both sides coalesced)
__global__ void k_incT(const float* __restrict__ inc, _Float16* __restrict__ incT) {
  __shared__ _Float16 tile[32][33];
  int blk = blockIdx.x;
  const int e0 = (blk % (E_ / 32)) * 32; blk /= (E_ / 32);
  const int m0 = (blk % (M_ / 32)) * 32; blk /= (M_ / 32);
  const int b  = blk;
  const int tx = threadIdx.x & 31, ty = threadIdx.x >> 5;   // 32 x 8
#pragma unroll
  for (int j = 0; j < 4; ++j) {
    int m = ty + j * 8;
    tile[m][tx] = (_Float16)inc[((size_t)(b * M_ + m0 + m)) * E_ + e0 + tx];
  }
  __syncthreads();
#pragma unroll
  for (int j = 0; j < 4; ++j) {
    int e = ty + j * 8;
    incT[((size_t)(b * E_ + e0 + e)) * M_ + m0 + tx] = tile[tx][e];
  }
}

// qkv16 [b,m,768] (V block) -> vt [b,h,dh,m] f16 via 32x32 LDS tiles
__global__ void k_vt(const _Float16* __restrict__ qkv, _Float16* __restrict__ vt) {
  __shared__ _Float16 tile[32][33];
  int blk = blockIdx.x;
  const int m0 = (blk % (M_ / 32)) * 32; blk /= (M_ / 32);
  const int h  = blk % H_;
  const int b  = blk / H_;
  const int tx = threadIdx.x & 31, ty = threadIdx.x >> 5;
#pragma unroll
  for (int j = 0; j < 4; ++j) {
    int m = ty + j * 8;   // rows = m, lanes = c (contiguous in qkv)
    tile[m][tx] = qkv[((size_t)(b * M_ + m0 + m)) * TD_ + 2 * D_ + h * DH_ + tx];
  }
  __syncthreads();
#pragma unroll
  for (int j = 0; j < 4; ++j) {
    int c = ty + j * 8;   // rows = c, lanes = m (contiguous in vt)
    vt[((size_t)((b * H_ + h) * DH_ + c)) * M_ + m0 + tx] = tile[tx][c];
  }
}

// ---------------- GEMM kernels: 128 threads = 4 waves per block -----------
// qkv = x @ Wqkv^T + bqkv ; x16 [B*M, D], w [3D, D] (K-major), out f16 [B*M,3D]
__global__ void k_gemm_qkv(const _Float16* __restrict__ x16,
                           const _Float16* __restrict__ w16,
                           const float* __restrict__ bias,
                           _Float16* __restrict__ qkv16) {
  const int grpN = TD_ / 64;                         // 12 groups of 4 tiles
  int wave = (blockIdx.x * blockDim.x + threadIdx.x) >> 5;
  int lane = threadIdx.x & 31;
  if (wave >= (B_ * M_ / 16) * grpN) return;
  int tm = wave / grpN, tn = wave % grpN;
  v8f acc[4] = {};
  gemm_tileN<4>(x16 + (size_t)tm * 16 * D_, D_,
                w16 + (size_t)tn * 64 * D_, D_, D_, acc);
  const int half = lane >> 4, idx = lane & 15;
#pragma unroll
  for (int t = 0; t < 4; ++t) {
    const int col = tn * 64 + t * 16 + idx;
    const float bv = bias[col];
#pragma unroll
    for (int r = 0; r < 8; ++r) {
      int row = tm * 16 + r + 8 * half;
      qkv16[(size_t)row * TD_ + col] = (_Float16)(acc[t][r] + bv);
    }
  }
}

// y = attn @ Wout^T + bout, stored transposed: yT [b, d, m] f16
__global__ void k_gemm_outproj(const _Float16* __restrict__ a16,
                               const _Float16* __restrict__ w16,
                               const float* __restrict__ bias,
                               _Float16* __restrict__ yT16) {
  const int grpN = D_ / 64;                          // 4
  int wave = (blockIdx.x * blockDim.x + threadIdx.x) >> 5;
  int lane = threadIdx.x & 31;
  if (wave >= (B_ * M_ / 16) * grpN) return;
  int tm = wave / grpN, tn = wave % grpN;
  v8f acc[4] = {};
  gemm_tileN<4>(a16 + (size_t)tm * 16 * D_, D_,
                w16 + (size_t)tn * 64 * D_, D_, D_, acc);
  const int half = lane >> 4, idx = lane & 15;
#pragma unroll
  for (int t = 0; t < 4; ++t) {
    const int col = tn * 64 + t * 16 + idx;
    const float bv = bias[col];
#pragma unroll
    for (int r = 0; r < 8; ++r) {
      int row  = tm * 16 + r + 8 * half;            // global row in [0, B*M)
      int b    = row >> 11;                         // /2048
      int mloc = row & (M_ - 1);
      yT16[((size_t)(b * D_ + col)) * M_ + mloc] = (_Float16)(acc[t][r] + bv);
    }
  }
}

// S[b,e,d] = sum_m incT[b,e,m] * yT[b,d,m]   (K = 2048, 2x64 WMMA per wave)
__global__ void k_gemm_S(const _Float16* __restrict__ incT,
                         const _Float16* __restrict__ yT,
                         float* __restrict__ S32) {
  const int tilesE = E_ / 16, grpD = D_ / 32;        // 64, 8
  int wave = (blockIdx.x * blockDim.x + threadIdx.x) >> 5;
  int lane = threadIdx.x & 31;
  if (wave >= B_ * tilesE * grpD) return;
  int t  = wave % (tilesE * grpD);
  int b  = wave / (tilesE * grpD);
  int te = t / grpD, td = t % grpD;
  v8f acc[2] = {};
  gemm_tileN<2>(incT + ((size_t)b * E_ + te * 16) * M_, M_,
                yT   + ((size_t)b * D_ + td * 32) * M_, M_, M_, acc);
  const int half = lane >> 4, idx = lane & 15;
#pragma unroll
  for (int t2 = 0; t2 < 2; ++t2)
#pragma unroll
    for (int r = 0; r < 8; ++r) {
      int e = te * 16 + r + 8 * half;
      S32[((size_t)(b * E_ + e)) * D_ + td * 32 + t2 * 16 + idx] = acc[t2][r];
    }
}

// p = ef @ Wproj^T ; ef16 [B*E, D], w [OUT, D], out f32 [B*E, OUT]
__global__ void k_gemm_proj(const _Float16* __restrict__ ef16,
                            const _Float16* __restrict__ w16,
                            float* __restrict__ p32) {
  const int grpN = OUT_ / 64;                        // 4
  int wave = (blockIdx.x * blockDim.x + threadIdx.x) >> 5;
  int lane = threadIdx.x & 31;
  if (wave >= (B_ * E_ / 16) * grpN) return;
  int tm = wave / grpN, tn = wave % grpN;
  v8f acc[4] = {};
  gemm_tileN<4>(ef16 + (size_t)tm * 16 * D_, D_,
                w16 + (size_t)tn * 64 * D_, D_, D_, acc);
  const int half = lane >> 4, idx = lane & 15;
#pragma unroll
  for (int t = 0; t < 4; ++t)
#pragma unroll
    for (int r = 0; r < 8; ++r) {
      int row = tm * 16 + r + 8 * half;
      p32[(size_t)row * OUT_ + tn * 64 + t * 16 + idx] = acc[t][r];
    }
}

// ---------------- flash attention (1 wave = 16 query rows of one head) ----
__global__ void k_attn(const _Float16* __restrict__ qkv16,
                       const _Float16* __restrict__ vt16,
                       _Float16* __restrict__ attn16) {
  __shared__ _Float16 Plds[8][16 * 32];              // per-wave 16x32 P tile
  const int wv   = threadIdx.x >> 5;
  const int lane = threadIdx.x & 31;
  const int half = lane >> 4, idx = lane & 15;
  int wave = blockIdx.x * (blockDim.x >> 5) + wv;
  const int QT = M_ / 16;                            // 128 query tiles
  if (wave >= B_ * H_ * QT) return;
  int qt = wave % QT;
  int bh = wave / QT;
  int h  = bh % H_;
  int b  = bh / H_;
  const int m0 = qt * 16;

  // Q fragment (A: 16x32, K = dh), loaded once
  HV qa;
  load_frag(qa, qkv16 + ((size_t)(b * M_ + m0 + idx)) * TD_ + h * DH_ + half * 8);

  float mrow[8], lrow[8];
#pragma unroll
  for (int r = 0; r < 8; ++r) { mrow[r] = -1e30f; lrow[r] = 0.f; }
  v8f O0 = {}, O1 = {};
  const float sc = 0.17677669529663687f;             // 1/sqrt(32)

  const _Float16* kbase = qkv16 + (size_t)b * M_ * TD_ + D_ + h * DH_;
  const _Float16* vbase = vt16 + ((size_t)(b * H_ + h)) * DH_ * M_;
  _Float16* pl = &Plds[wv][0];

  for (int kc = 0; kc < M_; kc += 32) {
    // ---- scores: two 16x16 tiles, one K=32 WMMA each
    HV kb0, kb1;
    const _Float16* k0r = kbase + (size_t)(kc + idx) * TD_ + half * 8;
    load_frag(kb0, k0r);
    load_frag(kb1, k0r + 16 * TD_);
    // V fragments for this chunk issued early to overlap with softmax VALU
    HV vb0, vb1;
    const _Float16* v0r = vbase + (size_t)idx * M_ + kc + half * 8;
    load_frag(vb0, v0r);
    load_frag(vb1, v0r + (size_t)16 * M_);

    v8f s0 = {}, s1 = {};
    s0 = wmma16(qa, kb0, s0);
    s1 = wmma16(qa, kb1, s1);

    // ---- online softmax (row m = r + 8*half, cols across the 16-lane half)
#pragma unroll
    for (int r = 0; r < 8; ++r) {
      float v0 = s0[r] * sc, v1 = s1[r] * sc;
      float mx = fmaxf(v0, v1);
      mx = fmaxf(mx, __shfl_xor(mx, 1, 32));
      mx = fmaxf(mx, __shfl_xor(mx, 2, 32));
      mx = fmaxf(mx, __shfl_xor(mx, 4, 32));
      mx = fmaxf(mx, __shfl_xor(mx, 8, 32));
      float mnew = fmaxf(mrow[r], mx);
      float al   = __expf(mrow[r] - mnew);
      float p0   = __expf(v0 - mnew);
      float p1   = __expf(v1 - mnew);
      float ps = p0 + p1;
      ps += __shfl_xor(ps, 1, 32);
      ps += __shfl_xor(ps, 2, 32);
      ps += __shfl_xor(ps, 4, 32);
      ps += __shfl_xor(ps, 8, 32);
      lrow[r] = lrow[r] * al + ps;
      mrow[r] = mnew;
      O0[r] *= al;
      O1[r] *= al;
      const int m = r + 8 * half;
      pl[m * 32 + idx]      = (_Float16)p0;
      pl[m * 32 + idx + 16] = (_Float16)p1;
    }
    // per-wave DS ops are in-order; wait + compiler barrier for the relayout
    asm volatile("s_wait_dscnt 0" ::: "memory");

    // ---- P re-read as an A fragment (K = 32 keys)
    HV pa;
    load_frag(pa, pl + idx * 32 + half * 8);
    asm volatile("" ::: "memory");

    // ---- P x V : two output column tiles
    O0 = wmma16(pa, vb0, O0);
    O1 = wmma16(pa, vb1, O1);
  }

  // ---- epilogue: normalize and store f16
#pragma unroll
  for (int r = 0; r < 8; ++r) {
    const int m = m0 + r + 8 * half;
    const float inv = 1.0f / lrow[r];
    _Float16* dst = attn16 + ((size_t)(b * M_ + m)) * D_ + h * DH_;
    dst[idx]      = (_Float16)(O0[r] * inv);
    dst[idx + 16] = (_Float16)(O1[r] * inv);
  }
}

// ---------------- softmax over hyperedge dim (axis=1) ----------------
__global__ void k_colstats(const float* __restrict__ S32,
                           float* __restrict__ cmax, float* __restrict__ csum) {
  int wave = (blockIdx.x * blockDim.x + threadIdx.x) >> 5;
  int lane = threadIdx.x & 31;
  if (wave >= B_ * D_) return;
  int b = wave / D_, dcol = wave % D_;
  const float* col = S32 + (size_t)b * E_ * D_ + dcol;
  float mx = -1e30f;
  for (int e = lane; e < E_; e += 32) mx = fmaxf(mx, col[(size_t)e * D_]);
  for (int msk = 16; msk; msk >>= 1) mx = fmaxf(mx, __shfl_xor(mx, msk, 32));
  float s = 0.f;
  for (int e = lane; e < E_; e += 32) s += __expf(col[(size_t)e * D_] - mx);
  for (int msk = 16; msk; msk >>= 1) s += __shfl_xor(s, msk, 32);
  if (lane == 0) { cmax[wave] = mx; csum[wave] = s; }
}

__global__ void k_ef(const float* __restrict__ S32,
                     const float* __restrict__ cmax, const float* __restrict__ csum,
                     _Float16* __restrict__ ef16) {
  int i = blockIdx.x * blockDim.x + threadIdx.x;
  if (i >= B_ * E_ * D_) return;
  int dcol = i % D_;
  int b    = i / (E_ * D_);
  float s = S32[i];
  float w = __expf(s - cmax[b * D_ + dcol]) / csum[b * D_ + dcol];
  ef16[i] = (_Float16)(s * w);
}

// ---------------- layernorm + residual (1 wave per row) ----------------
__global__ void k_ln(const float* __restrict__ p32, const float* __restrict__ prev,
                     const float* __restrict__ gamma, const float* __restrict__ beta,
                     const float* __restrict__ alphap, float* __restrict__ outp) {
  int wave = (blockIdx.x * blockDim.x + threadIdx.x) >> 5;
  int lane = threadIdx.x & 31;
  if (wave >= B_ * E_) return;
  const float* row = p32 + (size_t)wave * OUT_;
  float x[8], s = 0.f, s2 = 0.f;
#pragma unroll
  for (int j = 0; j < 8; ++j) {
    x[j] = row[j * 32 + lane];
    s += x[j]; s2 += x[j] * x[j];
  }
  for (int msk = 16; msk; msk >>= 1) {
    s  += __shfl_xor(s,  msk, 32);
    s2 += __shfl_xor(s2, msk, 32);
  }
  const float mu  = s * (1.0f / OUT_);
  const float var = s2 * (1.0f / OUT_) - mu * mu;
  const float rs  = rsqrtf(var + 1e-5f);
  const float alpha = alphap[0];
  const float* pr = prev + (size_t)wave * OUT_;
  float* op = outp + (size_t)wave * OUT_;
#pragma unroll
  for (int j = 0; j < 8; ++j) {
    int c = j * 32 + lane;
    float ln = (x[j] - mu) * rs * gamma[c] + beta[c];
    op[c] = alpha * pr[c] + (1.0f - alpha) * ln + pr[c];
  }
}

// ---------------- host launch ----------------
extern "C" void kernel_launch(void* const* d_in, const int* in_sizes, int n_in,
                              void* d_out, int out_size, void* d_ws, size_t ws_size,
                              hipStream_t stream) {
  const float* feature = (const float*)d_in[0];
  const float* inc     = (const float*)d_in[1];
  const float* prev    = (const float*)d_in[2];
  const float* wqkv    = (const float*)d_in[3];
  const float* bqkv    = (const float*)d_in[4];
  const float* wout    = (const float*)d_in[5];
  const float* bout    = (const float*)d_in[6];
  const float* wproj   = (const float*)d_in[7];
  const float* gamma   = (const float*)d_in[8];
  const float* beta    = (const float*)d_in[9];
  const float* alphap  = (const float*)d_in[10];
  float* outp = (float*)d_out;

  char* base = (char*)d_ws;
  size_t o = 0;
  auto carve = [&](size_t bytes) -> char* {
    char* p = base + o;
    o = (o + bytes + 255) & ~(size_t)255;
    return p;
  };
  _Float16* x16    = (_Float16*)carve((size_t)B_ * M_ * D_ * 2);
  _Float16* wqkv16 = (_Float16*)carve((size_t)TD_ * D_ * 2);
  _Float16* wout16 = (_Float16*)carve((size_t)D_ * D_ * 2);
  _Float16* wproj16= (_Float16*)carve((size_t)OUT_ * D_ * 2);
  _Float16* qkv16  = (_Float16*)carve((size_t)B_ * M_ * TD_ * 2);
  _Float16* vt16   = (_Float16*)carve((size_t)B_ * H_ * DH_ * M_ * 2);
  _Float16* attn16 = (_Float16*)carve((size_t)B_ * M_ * D_ * 2);
  _Float16* yT16   = (_Float16*)carve((size_t)B_ * D_ * M_ * 2);
  _Float16* incT16 = (_Float16*)carve((size_t)B_ * E_ * M_ * 2);
  float*    S32    = (float*)carve((size_t)B_ * E_ * D_ * 4);
  float*    cmax   = (float*)carve((size_t)B_ * D_ * 4);
  float*    csum   = (float*)carve((size_t)B_ * D_ * 4);
  _Float16* ef16   = (_Float16*)carve((size_t)B_ * E_ * D_ * 2);
  float*    p32    = (float*)carve((size_t)B_ * E_ * OUT_ * 4);

  const int T = 256;    // block for elementwise / attention kernels
  const int TG = 128;   // block for async-LDS GEMM kernels (4 waves)
  auto blk = [](int n, int t) { return (n + t - 1) / t; };

  // f32 -> f16 conversions / repacks
  k_f32_to_f16<<<blk(B_*M_*D_, T), T, 0, stream>>>(feature, x16, B_*M_*D_);
  k_f32_to_f16<<<blk(TD_*D_, T),  T, 0, stream>>>(wqkv, wqkv16, TD_*D_);
  k_f32_to_f16<<<blk(D_*D_, T),   T, 0, stream>>>(wout, wout16, D_*D_);
  k_f32_to_f16<<<blk(OUT_*D_, T), T, 0, stream>>>(wproj, wproj16, OUT_*D_);
  k_incT<<<B_ * (M_/32) * (E_/32), T, 0, stream>>>(inc, incT16);

  // QKV projection (WMMA), repack V
  {
    int waves = (B_*M_/16) * (TD_/64);
    k_gemm_qkv<<<blk(waves*32, TG), TG, 0, stream>>>(x16, wqkv16, bqkv, qkv16);
  }
  k_vt<<<B_ * H_ * (M_/32), T, 0, stream>>>(qkv16, vt16);

  // flash attention (WMMA scores + WMMA P·V)
  {
    int waves = B_ * H_ * (M_/16);
    k_attn<<<blk(waves*32, T), T, 0, stream>>>(qkv16, vt16, attn16);
  }

  // out projection -> transposed f16
  {
    int waves = (B_*M_/16) * (D_/64);
    k_gemm_outproj<<<blk(waves*32, TG), TG, 0, stream>>>(attn16, wout16, bout, yT16);
  }

  // S = inc^T @ attn_out  (K = 2048)
  {
    int waves = B_ * (E_/16) * (D_/32);
    k_gemm_S<<<blk(waves*32, TG), TG, 0, stream>>>(incT16, yT16, S32);
  }

  // softmax over hyperedges, ef = S * W
  k_colstats<<<blk(B_*D_*32, T), T, 0, stream>>>(S32, cmax, csum);
  k_ef<<<blk(B_*E_*D_, T), T, 0, stream>>>(S32, cmax, csum, ef16);

  // projection GEMM
  {
    int waves = (B_*E_/16) * (OUT_/64);
    k_gemm_proj<<<blk(waves*32, TG), TG, 0, stream>>>(ef16, wproj16, p32);
  }

  // layernorm + residual combine
  k_ln<<<blk(B_*E_*32, T), T, 0, stream>>>(p32, prev, gamma, beta, alphap, outp);
}
